// tdr_graph_constructor_35888746726124
// MI455X (gfx1250) — matrix operand order repaired and checked by reading, not compiled, not run on hardware
//
#include <hip/hip_runtime.h>

typedef _Float16 vh16 __attribute__((ext_vector_type(16)));
typedef float    vf8  __attribute__((ext_vector_type(8)));
typedef float    vf4  __attribute__((ext_vector_type(4)));

#define NNODES 2048
#define LAGMAX 4
#define KTOP   20
#define DIM    40
#define DIMP   64          // K padded to 64 for 2x K=32 WMMA
#define NCOL   8192        // N
#define ALPHA  3.0f
#define CSPLIT 8           // column splits for f1k occupancy

// Branch-free tanh: 1 - 2/(exp(2x)+1).
// exp(2x)->0 => -1 ; exp(2x)->inf => rcp(inf)=0 => +1. No NaN, no EXEC branching.
__device__ __forceinline__ float fast_tanhf(float x) {
    float e = __expf(2.0f * x);                     // v_exp_f32 path
    return 1.0f - 2.0f * __builtin_amdgcn_rcpf(e + 1.0f);
}

// ---------------------------------------------------------------------------
// Kernel Z: zero the whole 256MB output with nontemporal 128-bit stores.
// ---------------------------------------------------------------------------
__global__ __launch_bounds__(256) void zk(vf4* __restrict__ o) {
    size_t n = (size_t)NCOL * NCOL / 4;
    size_t stride = (size_t)gridDim.x * blockDim.x;
    vf4 z = {0.f, 0.f, 0.f, 0.f};
    for (size_t i = (size_t)blockIdx.x * blockDim.x + threadIdx.x; i < n; i += stride)
        __builtin_nontemporal_store(z, &o[i]);
}

// ---------------------------------------------------------------------------
// Kernel E: v2h[8192][64] = f16(tanh(3*(emb2[idx[r]] @ W2^T + b2))), K-padded,
//           v1h[2048][64] = f16(tanh(3*(emb1[idx[node*4+3]] @ W1^T + b1)))
// ---------------------------------------------------------------------------
__global__ __launch_bounds__(256) void ek(const int* __restrict__ idx,
                                          const float* __restrict__ emb1,
                                          const float* __restrict__ emb2,
                                          const float* __restrict__ w1,
                                          const float* __restrict__ b1,
                                          const float* __restrict__ w2,
                                          const float* __restrict__ b2,
                                          _Float16* __restrict__ v1h,
                                          _Float16* __restrict__ v2h) {
    int t = blockIdx.x * blockDim.x + threadIdx.x;
    const int total2 = NCOL * DIMP;            // v2 elements
    const int total1 = NNODES * DIMP;          // v1 elements
    if (t < total2) {
        int r = t / DIMP, d = t % DIMP;
        float out = 0.f;
        if (d < DIM) {
            const float* e = emb2 + (size_t)idx[r] * DIM;
            const float* w = w2 + d * DIM;
            float s = b2[d];
            #pragma unroll
            for (int k = 0; k < DIM; ++k) s = fmaf(e[k], w[k], s);
            out = fast_tanhf(ALPHA * s);
        }
        v2h[t] = (_Float16)out;
    } else if (t < total2 + total1) {
        int t1 = t - total2;
        int node = t1 / DIMP, d = t1 % DIMP;
        float out = 0.f;
        if (d < DIM) {
            int g = node * LAGMAX + (LAGMAX - 1);
            const float* e = emb1 + (size_t)idx[g] * DIM;
            const float* w = w1 + d * DIM;
            float s = b1[d];
            #pragma unroll
            for (int k = 0; k < DIM; ++k) s = fmaf(e[k], w[k], s);
            out = fast_tanhf(ALPHA * s);
        }
        v1h[t1] = (_Float16)out;
    }
}

// ---------------------------------------------------------------------------
// Kernel F1: WMMA GEMM [2048x64] x [64x8192] -> activation -> segment peaks.
// grid = (128 row-blocks, CSPLIT col-splits), 256 threads (8 waves).
// Block handles 16 node-rows x (8192/CSPLIT) cols; each wave owns a 16-col
// tile per 128-col chunk; 2 x v_wmma_f32_16x16x32_f16 per tile (K=64 padded).
// ---------------------------------------------------------------------------
__global__ __launch_bounds__(256) void f1k(const _Float16* __restrict__ v1h,
                                           const _Float16* __restrict__ v2h,
                                           const float* __restrict__ noise,
                                           float* __restrict__ pkval,
                                           unsigned char* __restrict__ pksub) {
    __shared__ float sT[16 * 128];   // activated (pre-noise) tile

    const int tid  = threadIdx.x;
    const int lane = tid & 31;
    const int wv   = tid >> 5;       // wave 0..7
    const int l15  = lane & 15;
    const int hi   = lane >> 4;      // lane group 0/1
    const int rowBase = blockIdx.x * 16;   // node rows

    // A fragments (16x32 f16 layout): lane holds M=l15,
    // element e -> K = kb*32 + hi*16 + e
    vh16 afrag[2];
    {
        const _Float16* ap = v1h + (size_t)(rowBase + l15) * DIMP;
        #pragma unroll
        for (int kb = 0; kb < 2; ++kb) {
            const _Float16* p0 = ap + kb * 32 + hi * 16;
            const _Float16* p1 = p0 + 8;
            vh16 f;
            #pragma unroll
            for (int e = 0; e < 8; ++e) { f[e] = p0[e]; f[8 + e] = p1[e]; }
            afrag[kb] = f;
        }
    }

    const int chunksPer = (NCOL / 128) / CSPLIT;          // 8
    const int chunk0 = blockIdx.y * chunksPer;

    for (int cc = 0; cc < chunksPer; ++cc) {
        const int chunk = chunk0 + cc;
        const int c0 = chunk * 128 + wv * 16;             // this wave's 16 columns

        // B fragments (32x16 f16 layout): lane holds N=l15,
        // element e -> K = kb*32 + hi*16 + e  => contiguous 32B loads.
        // Load both K-blocks up front so one wait covers both WMMAs.
        const _Float16* bp = v2h + (size_t)(c0 + l15) * DIMP + hi * 16;
        vh16 b0 = *(const vh16*)(bp);
        vh16 b1 = *(const vh16*)(bp + 32);

        vf8 acc = {0.f, 0.f, 0.f, 0.f, 0.f, 0.f, 0.f, 0.f};
        acc = __builtin_amdgcn_wmma_f32_16x16x32_f16(
            false, afrag[0], false, b0, (short)0, acc, false, false);
        acc = __builtin_amdgcn_wmma_f32_16x16x32_f16(
            false, afrag[1], false, b1, (short)0, acc, false, false);

        // D layout: VGPR j, lanes 0-15 -> M=j, lanes 16-31 -> M=j+8; N=l15
        #pragma unroll
        for (int j = 0; j < 8; ++j) {
            int m = j + 8 * hi;
            sT[m * 128 + wv * 16 + l15] = fmaxf(fast_tanhf(ALPHA * acc[j]), 0.f);
        }
        __syncthreads();

        // Segment peaks: 16 rows x 32 segments; noise added here with
        // coalesced float4 reads. First-occurrence argmax tie-break.
        #pragma unroll
        for (int t2 = 0; t2 < 2; ++t2) {
            int task = tid + t2 * 256;
            int m  = task >> 5;
            int sg = task & 31;
            int grow = (rowBase + m) * LAGMAX + (LAGMAX - 1);
            vf4 nz = *(const vf4*)(noise + (size_t)grow * NCOL + chunk * 128 + sg * 4);
            const float* p = &sT[m * 128 + sg * 4];
            float c0v = p[0] + 0.01f * nz.x;
            float c1v = p[1] + 0.01f * nz.y;
            float c2v = p[2] + 0.01f * nz.z;
            float c3v = p[3] + 0.01f * nz.w;
            float best = c0v; int bs = 0;
            if (c1v > best) { best = c1v; bs = 1; }
            if (c2v > best) { best = c2v; bs = 2; }
            if (c3v > best) { best = c3v; bs = 3; }
            size_t o = (size_t)(rowBase + m) * (NCOL / LAGMAX) + chunk * 32 + sg;
            pkval[o] = best;
            pksub[o] = (unsigned char)bs;
        }
        __syncthreads();
    }
}

// ---------------------------------------------------------------------------
// Kernel F2: per-row top-20 over 2048 segment peaks, scatter into output.
// One block (256 threads) per node row; peaks cached in LDS; 20 iterative
// max-reductions with smaller-index tie-break (matches jax.lax.top_k).
// ---------------------------------------------------------------------------
__global__ __launch_bounds__(256) void f2k(const float* __restrict__ pkval,
                                           const unsigned char* __restrict__ pksub,
                                           float* __restrict__ out) {
    __shared__ float pv[NCOL / LAGMAX];
    __shared__ unsigned char ps[NCOL / LAGMAX];
    __shared__ float rv[256];
    __shared__ int   ri[256];

    const int row = blockIdx.x;       // node index
    const int tid = threadIdx.x;
    const int nseg = NCOL / LAGMAX;   // 2048

    for (int i = tid; i < nseg; i += 256) {
        pv[i] = pkval[(size_t)row * nseg + i];
        ps[i] = pksub[(size_t)row * nseg + i];
    }
    __syncthreads();

    const size_t outRow = (size_t)(row * LAGMAX + (LAGMAX - 1)) * NCOL;

    for (int p = 0; p < KTOP; ++p) {
        float best = -1.0f; int bi = 1 << 30;
        for (int i = tid; i < nseg; i += 256) {
            float v = pv[i];
            if (v > best) { best = v; bi = i; }   // increasing i => ties keep lower idx
        }
        rv[tid] = best; ri[tid] = bi;
        __syncthreads();
        for (int s = 128; s > 0; s >>= 1) {
            if (tid < s) {
                float vo = rv[tid + s]; int io = ri[tid + s];
                if (vo > rv[tid] || (vo == rv[tid] && io < ri[tid])) {
                    rv[tid] = vo; ri[tid] = io;
                }
            }
            __syncthreads();
        }
        if (tid == 0) {
            int j = ri[0];
            out[outRow + (size_t)j * LAGMAX + ps[j]] = rv[0];
            pv[j] = -1.0f;    // remove from further rounds (all real vals >= 0)
        }
        __syncthreads();
    }
}

// ---------------------------------------------------------------------------
extern "C" void kernel_launch(void* const* d_in, const int* in_sizes, int n_in,
                              void* d_out, int out_size, void* d_ws, size_t ws_size,
                              hipStream_t stream) {
    const int*   idx   = (const int*)  d_in[0];
    const float* emb1  = (const float*)d_in[1];
    const float* emb2  = (const float*)d_in[2];
    const float* w1    = (const float*)d_in[3];
    const float* b1    = (const float*)d_in[4];
    const float* w2    = (const float*)d_in[5];
    const float* b2    = (const float*)d_in[6];
    const float* noise = (const float*)d_in[7];
    float* out = (float*)d_out;

    // Workspace layout (~21.3 MB total)
    char* ws = (char*)d_ws;
    _Float16* v1h = (_Float16*)ws;                                   // 2048*64*2   = 256 KB
    _Float16* v2h = (_Float16*)(ws + (size_t)NNODES * DIMP * 2);     // 8192*64*2   = 1 MB
    char* after = ws + (size_t)NNODES * DIMP * 2 + (size_t)NCOL * DIMP * 2;
    float* pkval = (float*)after;                                    // 2048*2048*4 = 16 MB
    unsigned char* pksub =
        (unsigned char*)(after + (size_t)NNODES * (NCOL / LAGMAX) * 4); // 4 MB

    // 1) zero 256 MB output
    zk<<<16384, 256, 0, stream>>>((vf4*)out);

    // 2) embeddings -> tanh(3*Linear), f16, K padded to 64
    int totalE = NCOL * DIMP + NNODES * DIMP;
    ek<<<(totalE + 255) / 256, 256, 0, stream>>>(idx, emb1, emb2, w1, b1, w2, b2,
                                                 v1h, v2h);

    // 3) WMMA GEMM + activation + noise + segment peaks
    f1k<<<dim3(NNODES / 16, CSPLIT), 256, 0, stream>>>(v1h, v2h, noise, pkval, pksub);

    // 4) per-row top-20 + scatter
    f2k<<<NNODES, 256, 0, stream>>>(pkval, pksub, out);
}
